// MultiScaleAttention_71725953843292
// MI455X (gfx1250) — matrix operand order repaired and checked by reading, not compiled, not run on hardware
//
#include <hip/hip_runtime.h>
#include <hip/hip_bf16.h>

typedef _Float16 v16h  __attribute__((ext_vector_type(16)));
typedef _Float16 v8h   __attribute__((ext_vector_type(8)));
typedef float    v8f   __attribute__((ext_vector_type(8)));
typedef unsigned int u32x4 __attribute__((ext_vector_type(4)));
typedef int      i32x8 __attribute__((ext_vector_type(8)));
typedef int      i32x4 __attribute__((ext_vector_type(4)));

#define NB   4      // batch
#define HW   4096   // 64*64
#define CQ   64     // Cr (QK channel dim)
#define CV   128    // C  (V channel dim)
#define BM   64     // query rows per block (16 per wave, 4 waves)
#define BN   64     // KV positions per iteration (double-buffered)

// ---------------------------------------------------------------------------
// TDM helper: 2D tile load Global -> LDS via Tensor Data Mover (gfx1250).
// Descriptor bitfields per CDNA5 ISA ch.8 (D# group0/group1); groups 2/3 and
// the trailing group zero (2D tensor, no iteration/gather, not in a cluster).
// 6-arg builtin: (uint32x4 g0, int32x8 g1, int32x4, int32x4, int32x8, i32 cpol)
// ---------------------------------------------------------------------------
static __device__ __forceinline__ unsigned lds_offset(const void* p) {
  // generic shared pointer: high 32 bits = LDS aperture, low 32 bits = LDS byte offset
  return (unsigned)(unsigned long long)(uintptr_t)p;
}

static __device__ __forceinline__ void tdm_load_2d(const _Float16* gptr, unsigned ldsoff,
                                                   unsigned tile_x, unsigned tile_y,
                                                   unsigned long long stride0_elems) {
  unsigned long long ga = (unsigned long long)(uintptr_t)gptr;
  u32x4 g0;
  g0[0] = 1u;                                               // count=1, user descriptor
  g0[1] = ldsoff;                                           // lds_addr
  g0[2] = (unsigned)(ga & 0xFFFFFFFFull);                   // global_addr[31:0]
  g0[3] = (unsigned)((ga >> 32) & 0x1FFFFFFull) | (2u << 30); // global_addr[56:32] | type=2
  i32x8 g1;
  g1[0] = (int)(1u << 16);                                  // workgroup_mask=0, data_size=1 (2B)
  g1[1] = (int)((tile_x & 0xFFFFu) << 16);                  // tensor_dim0[15:0] in bits 63:48
  g1[2] = (int)((tile_x >> 16) | ((tile_y & 0xFFFFu) << 16));   // tensor_dim0_hi | tensor_dim1_lo
  g1[3] = (int)(((tile_y >> 16) & 0xFFFFu) | ((tile_x & 0xFFFFu) << 16)); // tensor_dim1_hi | tile_dim0
  g1[4] = (int)(tile_y & 0xFFFFu);                          // tile_dim1 | tile_dim2=0
  g1[5] = (int)(stride0_elems & 0xFFFFFFFFull);             // tensor_dim0_stride[31:0]
  g1[6] = (int)((stride0_elems >> 32) & 0xFFFFull);         // stride0[47:32] | stride1_lo=0
  g1[7] = 0;
  i32x4 gz4 = {0, 0, 0, 0};
  i32x8 gz8 = {0, 0, 0, 0, 0, 0, 0, 0};
  __builtin_amdgcn_tensor_load_to_lds(g0, g1, gz4, gz4, gz8, 0);
}

// A-operand fragment (16x32 f16): lane=row, element e -> K=16*half+e
// -> two contiguous 8-half chunks at +16*half and +16*half+8.
static __device__ __forceinline__ v16h frag16(const _Float16* lo, const _Float16* hi) {
  v8h a = *(const v8h*)lo;
  v8h b = *(const v8h*)hi;
  return __builtin_shufflevector(a, b, 0,1,2,3,4,5,6,7,8,9,10,11,12,13,14,15);
}

// ---------------------------------------------------------------------------
// Stage 1: three 1x1 convs + PReLU, emit f16 in attention-friendly layouts.
//   e1 -> Qg[n][p][c]  (c<64)   e2 -> Kg[n][p][c]   asm -> Vg[n][c][p]
// ---------------------------------------------------------------------------
__global__ __launch_bounds__(256) void msa_conv1x1_prelu(
    const float* __restrict__ x,
    const float* __restrict__ w1, const float* __restrict__ b1, const float* __restrict__ a1,
    const float* __restrict__ w2, const float* __restrict__ b2, const float* __restrict__ a2,
    const float* __restrict__ wa, const float* __restrict__ ba, const float* __restrict__ aa,
    _Float16* __restrict__ Qg, _Float16* __restrict__ Kg, _Float16* __restrict__ Vg)
{
  const int t = blockIdx.x * blockDim.x + threadIdx.x;   // 0 .. NB*HW-1
  const int n = t >> 12;
  const int p = t & (HW - 1);
  const float* xcol = x + (size_t)n * CV * HW + p;
  const float s1 = a1[0], s2 = a2[0], sa = aa[0];

  for (int sel = 0; sel < 2; ++sel) {
    const float* w = sel ? w2 : w1;
    const float* b = sel ? b2 : b1;
    const float slope = sel ? s2 : s1;
    _Float16* dst = (sel ? Kg : Qg) + ((size_t)n * HW + p) * CQ;
    for (int o0 = 0; o0 < CQ; o0 += 16) {
      float acc[16];
      #pragma unroll
      for (int k = 0; k < 16; ++k) acc[k] = b[o0 + k];
      for (int c = 0; c < CV; ++c) {
        const float xc = xcol[(size_t)c * HW];
        #pragma unroll
        for (int k = 0; k < 16; ++k) acc[k] = fmaf(w[(o0 + k) * CV + c], xc, acc[k]);
      }
      #pragma unroll
      for (int k = 0; k < 16; ++k) {
        float y = acc[k];
        y = (y >= 0.0f) ? y : slope * y;
        dst[o0 + k] = (_Float16)y;
      }
    }
  }
  for (int o0 = 0; o0 < CV; o0 += 16) {
    float acc[16];
    #pragma unroll
    for (int k = 0; k < 16; ++k) acc[k] = ba[o0 + k];
    for (int c = 0; c < CV; ++c) {
      const float xc = xcol[(size_t)c * HW];
      #pragma unroll
      for (int k = 0; k < 16; ++k) acc[k] = fmaf(wa[(o0 + k) * CV + c], xc, acc[k]);
    }
    #pragma unroll
    for (int k = 0; k < 16; ++k) {
      float y = acc[k];
      y = (y >= 0.0f) ? y : sa * y;
      Vg[((size_t)n * CV + o0 + k) * HW + p] = (_Float16)y;
    }
  }
}

// ---------------------------------------------------------------------------
// Stage 2: flash attention, double-buffered TDM pipeline.
//   ST = K^T x Q tiles (softmax stats mostly in-lane), O^T = V x P^T.
//   4 waves/block; each wave owns 16 query columns i. TDM for tile t+1 is
//   issued before compute on tile t; s_wait_tensorcnt(2) retires tile t.
// ---------------------------------------------------------------------------
__global__ __launch_bounds__(128) void msa_flash_attention(
    const _Float16* __restrict__ Qg, const _Float16* __restrict__ Kg,
    const _Float16* __restrict__ Vg, float* __restrict__ out)
{
  __shared__ __attribute__((aligned(128))) _Float16 Qs[BM][CQ];       //  8 KB
  __shared__ __attribute__((aligned(128))) _Float16 Ks[2][BN][CQ];    // 16 KB
  __shared__ __attribute__((aligned(128))) _Float16 Vs[2][CV][BN];    // 32 KB

  const int n    = blockIdx.y;
  const int i0   = blockIdx.x * BM;
  const int tid  = threadIdx.x;
  const int wid  = tid >> 5;
  const int lane = tid & 31;
  const int lp   = lane & 15;
  const int h    = lane >> 4;

  const _Float16* Qbase = Qg + ((size_t)n * HW + i0) * CQ;
  const _Float16* Kbase = Kg + (size_t)n * HW * CQ;
  const _Float16* Vbase = Vg + (size_t)n * CV * HW;

  // Prologue: Q tile + first KV tile via TDM (wave 0 issues; TENSORcnt tracks).
  if (wid == 0) {
    tdm_load_2d(Qbase, lds_offset(&Qs[0][0]),    CQ, BM, CQ);
    tdm_load_2d(Kbase, lds_offset(&Ks[0][0][0]), CQ, BN, CQ);   // Ks[j][c]
    tdm_load_2d(Vbase, lds_offset(&Vs[0][0][0]), BN, CV, HW);   // Vs[c][j]
    __builtin_amdgcn_s_wait_tensorcnt(2);   // TDM in-order per wave: Q complete
  }
  __syncthreads();

  // Persistent Q B-fragments: B[K=c][N=i], lane=col i, element e -> c = 32*kc + 16*h + e
  v16h qb[2];
  {
    const int i = wid * 16 + lp;
    #pragma unroll
    for (int kc = 0; kc < 2; ++kc) {
      const _Float16* s = &Qs[i][kc * 32 + 16 * h];
      qb[kc] = frag16(s, s + 8);
    }
  }

  float mrow = -3.0e38f;   // running max (finite sentinel: avoids inf-inf NaN)
  float lrow = 0.0f;       // running denominator
  v8f  o[8];               // O^T accumulators: 8 channel groups of 16; lane=col i
  #pragma unroll
  for (int ct = 0; ct < 8; ++ct)
    #pragma unroll
    for (int r = 0; r < 8; ++r) o[ct][r] = 0.0f;

  const int NT = HW / BN;
  for (int it = 0; it < NT; ++it) {
    const int buf = it & 1;
    __syncthreads();                 // all waves done computing on buf^1 (tile it-1)
    if (wid == 0) {
      if (it + 1 < NT) {             // prefetch tile it+1 into the other buffer
        const int nb = buf ^ 1;
        tdm_load_2d(Kbase + (size_t)(it + 1) * BN * CQ, lds_offset(&Ks[nb][0][0]), CQ, BN, CQ);
        tdm_load_2d(Vbase + (size_t)(it + 1) * BN,      lds_offset(&Vs[nb][0][0]), BN, CV, HW);
        __builtin_amdgcn_s_wait_tensorcnt(2);   // tile it (buf) retired; it+1 in flight
      } else {
        __builtin_amdgcn_s_wait_tensorcnt(0);
      }
    }
    __syncthreads();                 // buf tile visible to all waves

    // ST tiles: ST[j,i] = sum_c K[c,j]*Q[c,i]; tile tt covers j in [16*tt, 16*tt+16)
    v8f st[4];
    #pragma unroll
    for (int tt = 0; tt < 4; ++tt)
      #pragma unroll
      for (int r = 0; r < 8; ++r) st[tt][r] = 0.0f;

    #pragma unroll
    for (int kc = 0; kc < 2; ++kc) {
      #pragma unroll
      for (int tt = 0; tt < 4; ++tt) {
        const _Float16* ka = &Ks[buf][tt * 16 + lp][kc * 32 + 16 * h];
        v16h af = frag16(ka, ka + 8);
        st[tt] = __builtin_amdgcn_wmma_f32_16x16x32_f16(
            false, af, false, qb[kc], (short)0, st[tt], false, false);
      }
    }

    // Online softmax per column i (lane owns col i; its regs hold 32 of 64 j's,
    // partner lane (xor 16) holds the other 32)
    float mc = st[0][0];
    #pragma unroll
    for (int tt = 0; tt < 4; ++tt)
      #pragma unroll
      for (int r = 0; r < 8; ++r) mc = fmaxf(mc, st[tt][r]);
    mc = fmaxf(mc, __shfl_xor(mc, 16, 32));
    const float mn  = fmaxf(mrow, mc);
    const float scl = __expf(mrow - mn);
    float pe[4][8], rs = 0.0f;
    #pragma unroll
    for (int tt = 0; tt < 4; ++tt)
      #pragma unroll
      for (int r = 0; r < 8; ++r) {
        pe[tt][r] = __expf(st[tt][r] - mn);
        rs += pe[tt][r];
      }
    rs += __shfl_xor(rs, 16, 32);
    lrow = lrow * scl + rs;
    mrow = mn;
    #pragma unroll
    for (int ct = 0; ct < 8; ++ct)
      #pragma unroll
      for (int r = 0; r < 8; ++r) o[ct][r] *= scl;

    // Assemble P^T B-operands: for j-chunk jc, element e -> j = 32*jc + 16*h + e,
    // i.e. tile 2*jc+h; rows 8..15 of that tile live in the partner half-lane.
    v16h pb[2];
    #pragma unroll
    for (int jc = 0; jc < 2; ++jc) {
      #pragma unroll
      for (int r = 0; r < 8; ++r) {
        const float own  = h ? pe[2 * jc + 1][r] : pe[2 * jc][r];      // this half's tile
        const float send = h ? pe[2 * jc][r]     : pe[2 * jc + 1][r];  // partner's tile
        const float recv = __shfl_xor(send, 16, 32);
        pb[jc][r]     = (_Float16)(h ? recv : own);    // e < 8  (rows 0..7)
        pb[jc][r + 8] = (_Float16)(h ? own  : recv);   // e >= 8 (rows 8..15)
      }
    }

    // O^T += V x P^T : A = V tile (16 c rows x 32 j), per channel group & j-chunk
    #pragma unroll
    for (int ct = 0; ct < 8; ++ct) {
      #pragma unroll
      for (int jc = 0; jc < 2; ++jc) {
        const _Float16* va = &Vs[buf][ct * 16 + lp][jc * 32 + 16 * h];
        v16h af = frag16(va, va + 8);
        o[ct] = __builtin_amdgcn_wmma_f32_16x16x32_f16(
            false, af, false, pb[jc], (short)0, o[ct], false, false);
      }
    }
  }

  // Epilogue: out[n][c][p], p = i0 + wid*16 + lp, c = ct*16 + r + 8*h
  const float rl = 1.0f / lrow;
  const int   ig = i0 + wid * 16 + lp;
  #pragma unroll
  for (int ct = 0; ct < 8; ++ct) {
    #pragma unroll
    for (int r = 0; r < 8; ++r) {
      const int c = ct * 16 + r + 8 * h;
      out[((size_t)n * CV + c) * HW + ig] = o[ct][r] * rl;
    }
  }
}

// ---------------------------------------------------------------------------
extern "C" void kernel_launch(void* const* d_in, const int* in_sizes, int n_in,
                              void* d_out, int out_size, void* d_ws, size_t ws_size,
                              hipStream_t stream) {
  const float* x  = (const float*)d_in[0];
  const float* w1 = (const float*)d_in[1];
  const float* b1 = (const float*)d_in[2];
  const float* a1 = (const float*)d_in[3];
  const float* w2 = (const float*)d_in[4];
  const float* b2 = (const float*)d_in[5];
  const float* a2 = (const float*)d_in[6];
  const float* wa = (const float*)d_in[7];
  const float* ba = (const float*)d_in[8];
  const float* aa = (const float*)d_in[9];
  float* out = (float*)d_out;

  // workspace: Qg 2MB | Kg 2MB | Vg 4MB (f16)
  _Float16* Qg = (_Float16*)d_ws;
  _Float16* Kg = Qg + (size_t)NB * HW * CQ;
  _Float16* Vg = Kg + (size_t)NB * HW * CQ;

  msa_conv1x1_prelu<<<(NB * HW) / 256, 256, 0, stream>>>(
      x, w1, b1, a1, w2, b2, a2, wa, ba, aa, Qg, Kg, Vg);

  dim3 grid(HW / BM, NB);
  msa_flash_attention<<<grid, 128, 0, stream>>>(Qg, Kg, Vg, out);
}
